// PureVQVAE_19782619365501
// MI455X (gfx1250) — matrix operand order repaired and checked by reading, not compiled, not run on hardware
//
#include <hip/hip_runtime.h>

typedef __attribute__((ext_vector_type(16))) _Float16 v16h;
typedef __attribute__((ext_vector_type(8)))  float    v8f;
typedef __attribute__((ext_vector_type(2)))  _Float16 h2;

#define LRELU_SLOPE 0.01f
#define BN_EPS 1e-5f

// ---------------- x [32][256][2048] f32 -> xt [32][2048][256] f16 (channels-last)
__global__ void x_transpose(const float* __restrict__ x, _Float16* __restrict__ xt) {
  __shared__ _Float16 tile[32][33];
  const int b = blockIdx.z;
  const int l0 = blockIdx.x * 32;
  const int c0 = blockIdx.y * 32;
  const int tx = threadIdx.x;  // 0..31 along L on read, along C on write
  const int ty = threadIdx.y;  // 0..7
#pragma unroll
  for (int j = 0; j < 4; ++j) {
    int c = c0 + ty + j * 8;
    tile[ty + j * 8][tx] = (_Float16)x[(size_t)(b * 256 + c) * 2048 + l0 + tx];
  }
  __syncthreads();
#pragma unroll
  for (int j = 0; j < 4; ++j) {
    int l = l0 + ty + j * 8;
    xt[(size_t)(b * 2048 + l) * 256 + c0 + tx] = tile[tx][ty + j * 8];
  }
}

// ---------------- weight repack: w[O][I][3] f32 -> wt[O][3*I] f16 (k = t*Cin + c)
__global__ void wt_prep(const float* __restrict__ w, _Float16* __restrict__ wt,
                        int Cin, int Cout) {
  int n = Cout * 3 * Cin;
  for (int idx = blockIdx.x * blockDim.x + threadIdx.x; idx < n;
       idx += gridDim.x * blockDim.x) {
    int o = idx / (3 * Cin);
    int rem = idx - o * 3 * Cin;
    int t = rem / Cin;
    int c = rem - t * Cin;
    wt[idx] = (_Float16)w[(o * Cin + c) * 3 + t];
  }
}

// ---------------- fold conv-bias + BN into per-channel alpha/beta
__global__ void bn_prep(const float* __restrict__ bias, const float* __restrict__ g,
                        const float* __restrict__ be, const float* __restrict__ m,
                        const float* __restrict__ var, float* __restrict__ ab, int C) {
  int i = blockIdx.x * blockDim.x + threadIdx.x;
  if (i < C) {
    float a = g[i] * rsqrtf(var[i] + BN_EPS);
    ab[i] = a;
    ab[C + i] = (bias[i] - m[i]) * a + be[i];
  }
}

// ---------------- emb f32 [4096][128] -> f16 copy + row ||e||^2
__global__ void emb_prep(const float* __restrict__ emb, _Float16* __restrict__ embh,
                         float* __restrict__ enorm) {
  __shared__ float red[128];
  int row = blockIdx.x, d = threadIdx.x;
  float v = emb[row * 128 + d];
  embh[row * 128 + d] = (_Float16)v;
  red[d] = v * v;
  __syncthreads();
  for (int s = 64; s; s >>= 1) {
    if (d < s) red[d] += red[d + s];
    __syncthreads();
  }
  if (d == 0) enorm[row] = red[0];
}

// ---------------- ||f||^2 per row of flat f16 [8192][128]
__global__ void fnorm_prep(const _Float16* __restrict__ flat, float* __restrict__ fnorm) {
  __shared__ float red[128];
  int row = blockIdx.x, d = threadIdx.x;
  float v = (float)flat[row * 128 + d];
  red[d] = v * v;
  __syncthreads();
  for (int s = 64; s; s >>= 1) {
    if (d < s) red[d] += red[d + s];
    __syncthreads();
  }
  if (d == 0) fnorm[row] = red[0];
}

// ---------------- implicit-GEMM conv (k=3, pad=1) + BN + LeakyReLU, WMMA f16
// Channels-last activations: in [B][Lin][Cin] f16, out [B][Lout][Cout] f16.
// One wave computes a 16-row x 64-col output tile.
// Every 32-wide K-chunk lies inside one tap (Cin % 32 == 0), so the im2col
// gather is contiguous per lane -> global_load_b128 pairs, like a plain GEMM.
// A (16x32): lane row m = lane&15; VGPR v -> K = 2(v&3)+16(v>=4)+8*(lane>>4)
// B (32x16): lane col n = lane&15; VGPR v -> K = 2(v&3)+16(v>=4)+8*(lane>>4)
// C (16x16): lane col n;           VGPR v -> row v + 8*(lane>>4)
__global__ void __launch_bounds__(32)
conv_gemm(const _Float16* __restrict__ act, const _Float16* __restrict__ wt,
          const float* __restrict__ ab, _Float16* __restrict__ out,
          int Cin, int Cout, int Lin, int Lout, int stride,
          int log2Lout, int shiftC) {
  const int lane = threadIdx.x;
  const int m = lane & 15, kh = lane >> 4;
  const int rowbase = blockIdx.x * 16;
  const int colbase = blockIdx.y * 64;
  const int K = 3 * Cin;

  const int r = rowbase + m;
  const int b = r >> log2Lout;
  const int lout = r & (Lout - 1);
  const int lin0 = lout * stride - 1;  // tap t -> lin0 + t

  v8f z = {};
  v8f acc[4];
#pragma unroll
  for (int j = 0; j < 4; ++j) acc[j] = z;

  for (int k0 = 0; k0 < K; k0 += 32) {
    const int t = k0 >> shiftC;       // uniform over the chunk
    const int c0 = k0 & (Cin - 1);
    const int lin = lin0 + t;
    const bool ok = (lin >= 0) && (lin < Lin);
    const int lc = lin < 0 ? 0 : (lin >= Lin ? Lin - 1 : lin);

    // ---- A fragment: contiguous channels-last gather
    const _Float16* ap = act + (size_t)(b * Lin + lc) * Cin + c0 + (kh << 3);
    const h2 pz = {};
    v16h a;
#pragma unroll
    for (int v = 0; v < 8; ++v) {
      h2 p = *(const h2*)(ap + 2 * (v & 3) + ((v >> 2) << 4));
      p = ok ? p : pz;
      a[2 * v] = p.x;
      a[2 * v + 1] = p.y;
    }
    // ---- 4 B fragments from wt[Cout][K] (contiguous 32-bit pairs)
#pragma unroll
    for (int j = 0; j < 4; ++j) {
      const _Float16* bp = wt + (size_t)(colbase + j * 16 + m) * K + k0 + (kh << 3);
      v16h bf;
#pragma unroll
      for (int v = 0; v < 8; ++v) {
        h2 p = *(const h2*)(bp + 2 * (v & 3) + ((v >> 2) << 4));
        bf[2 * v] = p.x;
        bf[2 * v + 1] = p.y;
      }
      acc[j] = __builtin_amdgcn_wmma_f32_16x16x32_f16(
          false, a, false, bf, (short)0, acc[j], false, false);
    }
  }

  // ---- epilogue: y = lrelu(acc*alpha + beta), channels-last store
#pragma unroll
  for (int j = 0; j < 4; ++j) {
    int o = colbase + j * 16 + m;
    float alpha = ab[o];
    float beta = ab[Cout + o];
#pragma unroll
    for (int v = 0; v < 8; ++v) {
      int rr = rowbase + v + 8 * kh;
      float y = acc[j][v] * alpha + beta;
      y = y >= 0.0f ? y : LRELU_SLOPE * y;
      out[(size_t)rr * Cout + o] = (_Float16)y;
    }
  }
}

// ---------------- distance GEMM + argmin
// Block = 256 (8 waves) owns 16 rows; wave w owns columns [w*512, w*512+512).
// dist = ||f||^2 + ||e||^2 - 2 * flat @ emb^T ; codes = argmin over 4096 cols.
__global__ void __launch_bounds__(256)
dist_gemm(const _Float16* __restrict__ flat, const _Float16* __restrict__ embh,
          const float* __restrict__ fnorm, const float* __restrict__ enorm,
          float* __restrict__ codes, float* __restrict__ dist) {
  __shared__ float smv[16][8];
  __shared__ int smi[16][8];
  const int tid = threadIdx.x;
  const int lane = tid & 31;
  const int wv = tid >> 5;
  const int m = lane & 15, kh = lane >> 4;
  const int rowbase = blockIdx.x * 16;

  // A fragments resident: flat rows rowbase..rowbase+15, K = 128 -> 4 frags
  v16h afr[4];
#pragma unroll
  for (int kt = 0; kt < 4; ++kt) {
    const _Float16* ap = flat + (size_t)(rowbase + m) * 128 + kt * 32 + (kh << 3);
    v16h a;
#pragma unroll
    for (int v = 0; v < 8; ++v) {
      h2 p = *(const h2*)(ap + 2 * (v & 3) + ((v >> 2) << 4));
      a[2 * v] = p.x;
      a[2 * v + 1] = p.y;
    }
    afr[kt] = a;
  }

  float fn[8];
#pragma unroll
  for (int v = 0; v < 8; ++v) fn[v] = fnorm[rowbase + v + 8 * kh];

  float minv[8];
  int mini[8];
#pragma unroll
  for (int v = 0; v < 8; ++v) {
    minv[v] = 3.4e38f;
    mini[v] = 0;
  }

  const int colstart = wv << 9;  // * 512
  for (int ct = 0; ct < 32; ++ct) {
    const int colbase = colstart + (ct << 4);
    if (ct + 1 < 32)  // prefetch next codebook tile (global_prefetch_b8)
      __builtin_prefetch(embh + (size_t)(colbase + 16 + m) * 128 + (kh << 6), 0, 0);

    v8f acc = {};
#pragma unroll
    for (int kt = 0; kt < 4; ++kt) {
      const _Float16* bp = embh + (size_t)(colbase + m) * 128 + kt * 32 + (kh << 3);
      v16h bf;
#pragma unroll
      for (int v = 0; v < 8; ++v) {
        h2 p = *(const h2*)(bp + 2 * (v & 3) + ((v >> 2) << 4));
        bf[2 * v] = p.x;
        bf[2 * v + 1] = p.y;
      }
      acc = __builtin_amdgcn_wmma_f32_16x16x32_f16(
          false, afr[kt], false, bf, (short)0, acc, false, false);
    }

    const int col = colbase + m;
    const float en = enorm[col];
#pragma unroll
    for (int v = 0; v < 8; ++v) {
      int rr = rowbase + v + 8 * kh;
      float dval = fn[v] + en - 2.0f * acc[v];
      dist[(size_t)rr * 4096 + col] = dval;
      if (dval < minv[v]) {
        minv[v] = dval;
        mini[v] = col;
      }
    }
  }

  // reduce argmin across the 16 lanes of each half-wave (wave32 shuffles)
#pragma unroll
  for (int v = 0; v < 8; ++v) {
    float mv = minv[v];
    int mi = mini[v];
#pragma unroll
    for (int off = 8; off > 0; off >>= 1) {
      float ov = __shfl_xor(mv, off, 32);
      int oi = __shfl_xor(mi, off, 32);
      if (ov < mv || (ov == mv && oi < mi)) {
        mv = ov;
        mi = oi;
      }
    }
    if (m == 0) {
      smv[v + 8 * kh][wv] = mv;
      smi[v + 8 * kh][wv] = mi;
    }
  }
  __syncthreads();
  if (tid < 16) {
    float best = smv[tid][0];
    int bi = smi[tid][0];
#pragma unroll
    for (int w = 1; w < 8; ++w) {
      float cv = smv[tid][w];
      int ci = smi[tid][w];
      if (cv < best || (cv == best && ci < bi)) {
        best = cv;
        bi = ci;
      }
    }
    codes[rowbase + tid] = (float)bi;
  }
}

extern "C" void kernel_launch(void* const* d_in, const int* in_sizes, int n_in,
                              void* d_out, int out_size, void* d_ws, size_t ws_size,
                              hipStream_t stream) {
  const float* x = (const float*)d_in[0];
  const float* w1 = (const float*)d_in[1];
  const float* b1 = (const float*)d_in[2];
  const float* w2 = (const float*)d_in[3];
  const float* b2 = (const float*)d_in[4];
  const float* w3 = (const float*)d_in[5];
  const float* b3 = (const float*)d_in[6];
  const float* w4 = (const float*)d_in[7];
  const float* b4 = (const float*)d_in[8];
  const float* g1 = (const float*)d_in[9];
  const float* be1 = (const float*)d_in[10];
  const float* m1 = (const float*)d_in[11];
  const float* v1 = (const float*)d_in[12];
  const float* g2 = (const float*)d_in[13];
  const float* be2 = (const float*)d_in[14];
  const float* m2 = (const float*)d_in[15];
  const float* v2 = (const float*)d_in[16];
  const float* g3 = (const float*)d_in[17];
  const float* be3 = (const float*)d_in[18];
  const float* m3 = (const float*)d_in[19];
  const float* v3 = (const float*)d_in[20];
  const float* g4 = (const float*)d_in[21];
  const float* be4 = (const float*)d_in[22];
  const float* m4 = (const float*)d_in[23];
  const float* v4 = (const float*)d_in[24];
  const float* emb = (const float*)d_in[25];

  (void)in_sizes; (void)n_in; (void)out_size; (void)ws_size;

  // bump-allocate workspace (256B aligned regions)
  char* base = (char*)d_ws;
  size_t off = 0;
  auto alloc = [&](size_t bytes) -> void* {
    void* p = base + off;
    off = (off + bytes + 255) & ~(size_t)255;
    return p;
  };

  _Float16* wt1 = (_Float16*)alloc((size_t)256 * 768 * 2);
  _Float16* wt2 = (_Float16*)alloc((size_t)512 * 768 * 2);
  _Float16* wt3 = (_Float16*)alloc((size_t)512 * 1536 * 2);
  _Float16* wt4 = (_Float16*)alloc((size_t)128 * 1536 * 2);
  float* ab1 = (float*)alloc(2 * 256 * 4);
  float* ab2 = (float*)alloc(2 * 512 * 4);
  float* ab3 = (float*)alloc(2 * 512 * 4);
  float* ab4 = (float*)alloc(2 * 128 * 4);
  _Float16* xt = (_Float16*)alloc((size_t)32 * 2048 * 256 * 2);   // channels-last x
  _Float16* act1 = (_Float16*)alloc((size_t)32 * 1024 * 256 * 2); // [B][L][C]
  _Float16* act2 = (_Float16*)alloc((size_t)32 * 512 * 512 * 2);
  _Float16* act3 = (_Float16*)alloc((size_t)32 * 256 * 512 * 2);
  _Float16* flat4 = (_Float16*)alloc((size_t)8192 * 128 * 2);     // [N][128]
  _Float16* embh = (_Float16*)alloc((size_t)4096 * 128 * 2);
  float* fnorm = (float*)alloc(8192 * 4);
  float* enorm = (float*)alloc(4096 * 4);

  float* codes = (float*)d_out;        // 8192 codes (as float)
  float* dist = (float*)d_out + 8192;  // 8192 x 4096 distances

  // ---- prep
  x_transpose<<<dim3(2048 / 32, 256 / 32, 32), dim3(32, 8), 0, stream>>>(x, xt);
  wt_prep<<<dim3((256 * 768 + 255) / 256), dim3(256), 0, stream>>>(w1, wt1, 256, 256);
  wt_prep<<<dim3((512 * 768 + 255) / 256), dim3(256), 0, stream>>>(w2, wt2, 256, 512);
  wt_prep<<<dim3((512 * 1536 + 255) / 256), dim3(256), 0, stream>>>(w3, wt3, 512, 512);
  wt_prep<<<dim3((128 * 1536 + 255) / 256), dim3(256), 0, stream>>>(w4, wt4, 512, 128);
  bn_prep<<<dim3(1), dim3(256), 0, stream>>>(b1, g1, be1, m1, v1, ab1, 256);
  bn_prep<<<dim3(2), dim3(256), 0, stream>>>(b2, g2, be2, m2, v2, ab2, 512);
  bn_prep<<<dim3(2), dim3(256), 0, stream>>>(b3, g3, be3, m3, v3, ab3, 512);
  bn_prep<<<dim3(1), dim3(256), 0, stream>>>(b4, g4, be4, m4, v4, ab4, 128);
  emb_prep<<<dim3(4096), dim3(128), 0, stream>>>(emb, embh, enorm);

  // ---- conv stack (implicit GEMM, WMMA f16, channels-last)
  // L1: 256->256, Lin 2048 -> Lout 1024, stride 2, N = 32768
  conv_gemm<<<dim3(32768 / 16, 256 / 64), dim3(32), 0, stream>>>(
      xt, wt1, ab1, act1, 256, 256, 2048, 1024, 2, 10, 8);
  // L2: 256->512, 1024 -> 512, stride 2, N = 16384
  conv_gemm<<<dim3(16384 / 16, 512 / 64), dim3(32), 0, stream>>>(
      act1, wt2, ab2, act2, 256, 512, 1024, 512, 2, 9, 8);
  // L3: 512->512, 512 -> 256, stride 2, N = 8192
  conv_gemm<<<dim3(8192 / 16, 512 / 64), dim3(32), 0, stream>>>(
      act2, wt3, ab3, act3, 512, 512, 512, 256, 2, 8, 9);
  // L4: 512->128, 256 -> 256, stride 1, N = 8192 -> flat [8192][128]
  conv_gemm<<<dim3(8192 / 16, 128 / 64), dim3(32), 0, stream>>>(
      act3, wt4, ab4, flat4, 512, 128, 256, 256, 1, 8, 9);

  // ---- norms + distance GEMM + argmin
  fnorm_prep<<<dim3(8192), dim3(128), 0, stream>>>(flat4, fnorm);
  dist_gemm<<<dim3(8192 / 16), dim3(256), 0, stream>>>(flat4, embh, fnorm, enorm,
                                                       codes, dist);
}
